// LCNetBlock_22754736735015
// MI455X (gfx1250) — matrix operand order repaired and checked by reading, not compiled, not run on hardware
//
#include <hip/hip_runtime.h>

// ---------------------------------------------------------------------------
// CDNA5 (gfx1250) WMMA types and helpers
// ---------------------------------------------------------------------------
typedef __attribute__((ext_vector_type(16))) _Float16 v16h;
typedef __attribute__((ext_vector_type(8)))  float    v8f;

__device__ __forceinline__ v8f wmma16(v16h a, v16h b, v8f c) {
  // v_wmma_f32_16x16x32_f16: D = A(16x32) * B(32x16) + C(16x16 f32)
  return __builtin_amdgcn_wmma_f32_16x16x32_f16(
      /*neg_a=*/false, a, /*neg_b=*/false, b,
      /*c_mod=*/(short)0, c, /*reuse_a=*/false, /*reuse_b=*/false);
}

// A-fragment (16x32 f16): lane l -> row m0+(l&15); lane-half selects 16-wide K chunk.
// V0..7: K = h*16 + 2v   (h = l>>4)
__device__ __forceinline__ v16h frag_a(const _Float16* t, int ldt, int m0,
                                       int kb, int lane) {
  const _Float16* row = t + (size_t)(m0 + (lane & 15)) * ldt + kb;
  int h = (lane >> 4) * 16;
  v16h a;
#pragma unroll
  for (int v = 0; v < 4; ++v) {
    a[2 * v]     = row[h + 2 * v];
    a[2 * v + 1] = row[h + 2 * v + 1];
  }
#pragma unroll
  for (int v = 0; v < 4; ++v) {
    a[8 + 2 * v] = row[h + 8 + 2 * v];
    a[9 + 2 * v] = row[h + 8 + 2 * v + 1];
  }
  return a;
}

// B-fragment from tile stored [N][K] (n-major): lane l -> col n0+(l&15);
// lane-half selects 16-wide K chunk; V0..7: K = h*16 + 2v
__device__ __forceinline__ v16h frag_b_nk(const _Float16* t, int ldt, int n0,
                                          int kb, int lane) {
  const _Float16* row = t + (size_t)(n0 + (lane & 15)) * ldt + kb;
  int h = (lane >> 4) * 16;
  v16h b;
#pragma unroll
  for (int v = 0; v < 8; ++v) {
    b[2 * v]     = row[h + 2 * v];
    b[2 * v + 1] = row[h + 2 * v + 1];
  }
  return b;
}

// B-fragment from tile stored [K][N] (k-major), e.g. a V block.
__device__ __forceinline__ v16h frag_b_kn(const _Float16* t, int ldt, int n0,
                                          int kb, int lane) {
  int n = n0 + (lane & 15);
  int h = (lane >> 4) * 16;
  v16h b;
#pragma unroll
  for (int v = 0; v < 8; ++v) {
    b[2 * v]     = t[(size_t)(kb + h + 2 * v) * ldt + n];
    b[2 * v + 1] = t[(size_t)(kb + h + 2 * v + 1) * ldt + n];
  }
  return b;
}

// ---------------------------------------------------------------------------
// Generic WMMA GEMM: Out[M,N] = A[M,K] * B[N,K]^T (+epilogue)
//   transA: A stored [K,M];  transB: B stored [K,N]
//   mode 0: plain, 1: +bias[n], 2: relu(x*g[n]*rsqrt(1+eps)+b[n])
//   transOut: store Out[n*ldo+m] else Out[m*ldo+n]; outF fp32 or outH fp16
// ---------------------------------------------------------------------------
__global__ __launch_bounds__(256) void gemm_wmma_kernel(
    const _Float16* __restrict__ A, int lda, int transA,
    const _Float16* __restrict__ Bm, int ldb, int transB,
    float* __restrict__ outF, _Float16* __restrict__ outH, int ldo, int transOut,
    int M, int N, int K,
    const float* __restrict__ bias,
    const float* __restrict__ bnG, const float* __restrict__ bnB, int mode) {
  __shared__ _Float16 As[128][40];
  __shared__ _Float16 Bs[64][40];
  int tid = threadIdx.x, lane = tid & 31, w = tid >> 5;
  int wm = w & 3, wn = w >> 2;           // 4 x 2 waves -> 128 x 64 tile
  int am0 = blockIdx.x * 128, bn0 = blockIdx.y * 64;
  v8f acc[2][2] = {};
  for (int k0 = 0; k0 < K; k0 += 32) {
#pragma unroll
    for (int i = 0; i < 16; ++i) {       // A tile: 128x32
      int e = i * 256 + tid;
      int r = e >> 5, k = e & 31;
      int gm = am0 + r, gk = k0 + k;
      float v = 0.f;
      if (gm < M && gk < K)
        v = (float)(transA ? A[(size_t)gk * lda + gm]
                           : A[(size_t)gm * lda + gk]);
      As[r][k] = (_Float16)v;
    }
#pragma unroll
    for (int i = 0; i < 8; ++i) {        // B tile: 64x32
      int e = i * 256 + tid;
      int r = e >> 5, k = e & 31;
      int gn = bn0 + r, gk = k0 + k;
      float v = 0.f;
      if (gn < N && gk < K)
        v = (float)(transB ? Bm[(size_t)gk * ldb + gn]
                           : Bm[(size_t)gn * ldb + gk]);
      Bs[r][k] = (_Float16)v;
    }
    if (!transA && (k0 + 32) < K) {      // hint next A tile (global_prefetch_b8)
      int gm = am0 + (tid >> 1);
      if (gm < M) __builtin_prefetch(&A[(size_t)gm * lda + k0 + 32], 0, 1);
    }
    __syncthreads();
    v16h af[2], bf[2];
#pragma unroll
    for (int s = 0; s < 2; ++s) af[s] = frag_a(&As[0][0], 40, wm * 32 + s * 16, 0, lane);
#pragma unroll
    for (int s = 0; s < 2; ++s) bf[s] = frag_b_nk(&Bs[0][0], 40, wn * 32 + s * 16, 0, lane);
#pragma unroll
    for (int i = 0; i < 2; ++i)
#pragma unroll
      for (int j = 0; j < 2; ++j) acc[i][j] = wmma16(af[i], bf[j], acc[i][j]);
    __syncthreads();
  }
#pragma unroll
  for (int i = 0; i < 2; ++i)
#pragma unroll
    for (int j = 0; j < 2; ++j)
#pragma unroll
      for (int r = 0; r < 8; ++r) {
        int m = am0 + wm * 32 + i * 16 + r + ((lane >> 4) * 8);
        int n = bn0 + wn * 32 + j * 16 + (lane & 15);
        if (m < M && n < N) {
          float v = acc[i][j][r];
          if (mode == 1) {
            if (bias) v += bias[n];
          } else if (mode == 2) {
            float s = bnG[n] * rsqrtf(1.f + 1e-5f);
            v = fmaxf(v * s + bnB[n], 0.f);
          }
          size_t idx = transOut ? ((size_t)n * ldo + m) : ((size_t)m * ldo + n);
          if (outF) outF[idx] = v;
          else      outH[idx] = (_Float16)v;
        }
      }
}

// ---------------------------------------------------------------------------
// Streaming attention (64-row blocks, WMMA scores + WMMA P*V)
// ---------------------------------------------------------------------------
__device__ __forceinline__ void attn_load64(const _Float16* g, _Float16* s, int tid) {
  for (int e = tid; e < 64 * 64; e += 256)
    s[(e >> 6) * 72 + (e & 63)] = g[(size_t)(e >> 6) * 64 + (e & 63)];
}

// S(64x64) = Qs * Ks^T * scale -> Ss (ld 68). 8 waves arranged 2(m) x 4(n=16).
__device__ __forceinline__ void attn_score(const _Float16* Qs, const _Float16* Ks,
                                           float* Ss, int lane, int wm, int wn,
                                           float scale) {
  v8f acc[2] = {};
#pragma unroll
  for (int kb = 0; kb < 64; kb += 32) {
    v16h bfr = frag_b_nk(Ks, 72, wn * 16, kb, lane);
#pragma unroll
    for (int s = 0; s < 2; ++s) {
      v16h afr = frag_a(Qs, 72, wm * 32 + s * 16, kb, lane);
      acc[s] = wmma16(afr, bfr, acc[s]);
    }
  }
#pragma unroll
  for (int s = 0; s < 2; ++s)
#pragma unroll
    for (int r = 0; r < 8; ++r) {
      int m = wm * 32 + s * 16 + r + ((lane >> 4) * 8);
      int n = wn * 16 + (lane & 15);
      Ss[m * 68 + n] = acc[s][r] * scale;
    }
}

__global__ __launch_bounds__(256) void attn_pass1_kernel(
    const _Float16* __restrict__ Q, const _Float16* __restrict__ Km,
    float* __restrict__ rowMax, float* __restrict__ rowSum, int Nn) {
  __shared__ _Float16 Qs[64][72];
  __shared__ _Float16 Ks[64][72];
  __shared__ float Ss[64][68];
  __shared__ float red[64][4];
  __shared__ float rmS[64];
  int b = blockIdx.y, i0 = blockIdx.x * 64;
  const _Float16* Qb = Q + ((size_t)b * Nn + i0) * 64;
  const _Float16* Kb = Km + (size_t)b * Nn * 64;
  int tid = threadIdx.x, lane = tid & 31, w = tid >> 5;
  int wm = w & 1, wn = w >> 1;
  attn_load64(Qb, &Qs[0][0], tid);
  int row = tid >> 2, seg = tid & 3;
  float lmax = -3.0e38f;
  for (int j0 = 0; j0 < Nn; j0 += 64) {
    __syncthreads();
    attn_load64(Kb + (size_t)j0 * 64, &Ks[0][0], tid);
    __syncthreads();
    attn_score(&Qs[0][0], &Ks[0][0], &Ss[0][0], lane, wm, wn, 0.125f);
    __syncthreads();
#pragma unroll
    for (int c = 0; c < 16; ++c) lmax = fmaxf(lmax, Ss[row][seg * 16 + c]);
  }
  red[row][seg] = lmax;
  __syncthreads();
  if (tid < 64) {
    float m4 = fmaxf(fmaxf(red[tid][0], red[tid][1]), fmaxf(red[tid][2], red[tid][3]));
    rmS[tid] = m4;
    rowMax[(size_t)b * Nn + i0 + tid] = m4;
  }
  __syncthreads();
  float rm = rmS[row];
  float lsum = 0.f;
  for (int j0 = 0; j0 < Nn; j0 += 64) {
    __syncthreads();
    attn_load64(Kb + (size_t)j0 * 64, &Ks[0][0], tid);
    __syncthreads();
    attn_score(&Qs[0][0], &Ks[0][0], &Ss[0][0], lane, wm, wn, 0.125f);
    __syncthreads();
#pragma unroll
    for (int c = 0; c < 16; ++c) lsum += __expf(Ss[row][seg * 16 + c] - rm);
  }
  __syncthreads();
  red[row][seg] = lsum;
  __syncthreads();
  if (tid < 64)
    rowSum[(size_t)b * Nn + i0 + tid] = red[tid][0] + red[tid][1] + red[tid][2] + red[tid][3];
}

__global__ __launch_bounds__(256) void attn_colsum_kernel(
    const _Float16* __restrict__ Q, const _Float16* __restrict__ Km,
    const float* __restrict__ rowMax, const float* __restrict__ rowSum,
    float* __restrict__ keyImp, int Nn) {
  __shared__ _Float16 Qs[64][72];
  __shared__ _Float16 Ks[64][72];
  __shared__ float Ss[64][68];
  __shared__ float red[64][4];
  __shared__ float rmS[64], zS[64];
  int b = blockIdx.y, i0 = blockIdx.x * 64;
  const _Float16* Qb = Q + ((size_t)b * Nn + i0) * 64;
  const _Float16* Kb = Km + (size_t)b * Nn * 64;
  int tid = threadIdx.x, lane = tid & 31, w = tid >> 5;
  int wm = w & 1, wn = w >> 1;
  if (tid < 64) {
    rmS[tid] = rowMax[(size_t)b * Nn + i0 + tid];
    zS[tid]  = rowSum[(size_t)b * Nn + i0 + tid];
  }
  attn_load64(Qb, &Qs[0][0], tid);
  int col = tid & 63, q = tid >> 6;
  for (int j0 = 0; j0 < Nn; j0 += 64) {
    __syncthreads();
    attn_load64(Kb + (size_t)j0 * 64, &Ks[0][0], tid);
    __syncthreads();
    attn_score(&Qs[0][0], &Ks[0][0], &Ss[0][0], lane, wm, wn, 0.125f);
    __syncthreads();
    float part = 0.f;
#pragma unroll
    for (int r = 0; r < 16; ++r) {
      int rw = q * 16 + r;
      part += __expf(Ss[rw][col] - rmS[rw]) / zS[rw];
    }
    red[col][q] = part;
    __syncthreads();
    if (tid < 64)
      atomicAdd(&keyImp[(size_t)b * Nn + j0 + tid],
                red[tid][0] + red[tid][1] + red[tid][2] + red[tid][3]);
  }
}

// Bitonic select of top-kv threshold -> binary mask km
__global__ __launch_bounds__(1024) void topk_mask_kernel(
    const float* __restrict__ keyImp, float* __restrict__ km, int Nn, int kv) {
  __shared__ float v[4096];
  int b = blockIdx.x, tid = threadIdx.x;
  const float* src = keyImp + (size_t)b * Nn;
  for (int i = tid; i < Nn; i += 1024) v[i] = src[i];
  __syncthreads();
  for (int k = 2; k <= Nn; k <<= 1)
    for (int j = k >> 1; j > 0; j >>= 1) {
      for (int i = tid; i < Nn; i += 1024) {
        int ixj = i ^ j;
        if (ixj > i) {
          float a = v[i], c = v[ixj];
          if ((a > c) == ((i & k) == 0)) { v[i] = c; v[ixj] = a; }  // ascending
        }
      }
      __syncthreads();
    }
  float thr = v[Nn - kv];
  __syncthreads();
  for (int i = tid; i < Nn; i += 1024)
    km[(size_t)b * Nn + i] = (src[i] >= thr) ? 1.f : 0.f;
}

__global__ __launch_bounds__(256) void attn_pass2_kernel(
    const _Float16* __restrict__ Q, const _Float16* __restrict__ Km,
    const _Float16* __restrict__ V,
    const float* __restrict__ rowMax, const float* __restrict__ rowSum,
    const float* __restrict__ kmask, _Float16* __restrict__ outA, int Nn) {
  __shared__ _Float16 Qs[64][72];
  __shared__ _Float16 Ks[64][72];
  __shared__ float Ss[64][68];     // aliased as V tile (f16, ld 100) later
  __shared__ _Float16 Ps[64][72];
  __shared__ float red[64][4];
  __shared__ float rmS[64], zS[64], msum[64];
  int b = blockIdx.y, i0 = blockIdx.x * 64;
  const _Float16* Qb = Q + ((size_t)b * Nn + i0) * 64;
  const _Float16* Kb = Km + (size_t)b * Nn * 64;
  const _Float16* Vb = V + (size_t)b * Nn * 96;
  int tid = threadIdx.x, lane = tid & 31, w = tid >> 5;
  int wm = w & 1, wn = w >> 1;      // score mapping: 2 x 4 (n groups of 16)
  int wmO = w & 1, wnO = w >> 1;    // output mapping: 2 x 4 (n groups of 32), wnO<3 active
  _Float16* Vsp = reinterpret_cast<_Float16*>(&Ss[0][0]);
  if (tid < 64) {
    rmS[tid] = rowMax[(size_t)b * Nn + i0 + tid];
    zS[tid]  = rowSum[(size_t)b * Nn + i0 + tid];
  }
  attn_load64(Qb, &Qs[0][0], tid);
  int row = tid >> 2, seg = tid & 3;
  float lms = 0.f;
  v8f accO[2][2] = {};
  for (int j0 = 0; j0 < Nn; j0 += 64) {
    __syncthreads();
    attn_load64(Kb + (size_t)j0 * 64, &Ks[0][0], tid);
    __syncthreads();
    attn_score(&Qs[0][0], &Ks[0][0], &Ss[0][0], lane, wm, wn, 0.125f);
    __syncthreads();
    // masked, Z-normalized P
#pragma unroll
    for (int c = 0; c < 16; ++c) {
      int col = seg * 16 + c;
      float p = __expf(Ss[row][col] - rmS[row]) / zS[row] *
                kmask[(size_t)b * Nn + j0 + col];
      lms += p;
      Ps[row][col] = (_Float16)p;
    }
    __syncthreads();
    // V tile (64 x 96) into aliased LDS, k-major ld 100
    for (int e = tid; e < 64 * 96; e += 256)
      Vsp[(e / 96) * 100 + (e % 96)] = Vb[(size_t)(j0 + e / 96) * 96 + (e % 96)];
    __syncthreads();
    if (wnO < 3) {
#pragma unroll
      for (int kb = 0; kb < 64; kb += 32) {
        v16h bf0 = frag_b_kn(Vsp, 100, wnO * 32, kb, lane);
        v16h bf1 = frag_b_kn(Vsp, 100, wnO * 32 + 16, kb, lane);
#pragma unroll
        for (int s = 0; s < 2; ++s) {
          v16h af = frag_a(&Ps[0][0], 72, wmO * 32 + s * 16, kb, lane);
          accO[s][0] = wmma16(af, bf0, accO[s][0]);
          accO[s][1] = wmma16(af, bf1, accO[s][1]);
        }
      }
    }
  }
  __syncthreads();
  red[row][seg] = lms;
  __syncthreads();
  if (tid < 64)
    msum[tid] = red[tid][0] + red[tid][1] + red[tid][2] + red[tid][3] + 1e-8f;
  __syncthreads();
  if (wnO < 3) {
#pragma unroll
    for (int si = 0; si < 2; ++si)
#pragma unroll
      for (int sj = 0; sj < 2; ++sj)
#pragma unroll
        for (int r = 0; r < 8; ++r) {
          int m = wmO * 32 + si * 16 + r + ((lane >> 4) * 8);
          int n = wnO * 32 + sj * 16 + (lane & 15);
          outA[((size_t)b * Nn + i0 + m) * 96 + n] =
              (_Float16)(accO[si][sj][r] / msum[m]);
        }
  }
}

// ---------------------------------------------------------------------------
// Scalar helper kernels (bandwidth-trivial pieces)
// ---------------------------------------------------------------------------
__global__ void dwconv_kernel(const float* __restrict__ in, float* __restrict__ out,
                              const float* __restrict__ wgt, const float* __restrict__ bias,
                              const float* __restrict__ g, const float* __restrict__ bb,
                              const float* __restrict__ resid,
                              int H, int W, int ks, int mode) {
  int c = blockIdx.y;
  int p = blockIdx.x * blockDim.x + threadIdx.x;
  if (p >= H * W) return;
  int y = p / W, x = p % W, pad = ks / 2;
  const float* wc = wgt + (size_t)c * ks * ks;
  const float* ic = in + (size_t)c * H * W;
  float acc = 0.f;
  for (int ky = 0; ky < ks; ++ky) {
    int yy = y + ky - pad;
    if (yy < 0 || yy >= H) continue;
    for (int kx = 0; kx < ks; ++kx) {
      int xx = x + kx - pad;
      if (xx < 0 || xx >= W) continue;
      acc += wc[ky * ks + kx] * ic[yy * W + xx];
    }
  }
  if (bias) acc += bias[c];
  if (mode == 1) {                       // relu(bn(x))
    acc = fmaxf(acc * (g[c] * rsqrtf(1.f + 1e-5f)) + bb[c], 0.f);
  } else if (mode == 2) {                // bn(relu(x))
    acc = fmaxf(acc, 0.f);
    acc = acc * (g[c] * rsqrtf(1.f + 1e-5f)) + bb[c];
  } else if (mode == 3) {                // bn(x)
    acc = acc * (g[c] * rsqrtf(1.f + 1e-5f)) + bb[c];
  }
  if (resid) acc += resid[(size_t)c * H * W + p];
  out[(size_t)c * H * W + p] = acc;
}

__global__ void cast_f16_kernel(const float* __restrict__ s, _Float16* __restrict__ d, int n) {
  int i = blockIdx.x * blockDim.x + threadIdx.x;
  if (i < n) d[i] = (_Float16)s[i];
}

__global__ void zero_kernel(float* __restrict__ p, int n) {
  int i = blockIdx.x * blockDim.x + threadIdx.x;
  if (i < n) p[i] = 0.f;
}

__global__ void im2col3_kernel(const float* __restrict__ src, _Float16* __restrict__ col,
                               int C, int H, int W) {
  int e = blockIdx.x * blockDim.x + threadIdx.x;
  int K = C * 9;
  if (e >= H * W * K) return;
  int m = e / K, k = e % K;
  int c = k / 9, r = k % 9;
  int y = m / W + r / 3 - 1, x = m % W + r % 3 - 1;
  float v = 0.f;
  if (y >= 0 && y < H && x >= 0 && x < W) v = src[((size_t)c * H + y) * W + x];
  col[e] = (_Float16)v;
}

__global__ void gather_f_kernel(const float* __restrict__ f1, const float* __restrict__ f3,
                                const float* __restrict__ f5, _Float16* __restrict__ f,
                                int C, int HW) {
  int b = blockIdx.y;
  int idx = blockIdx.x * blockDim.x + threadIdx.x;
  if (idx >= HW * 3 * C) return;
  int n = idx / (3 * C), c = idx % (3 * C);
  const float* s = (c < C) ? f1 : (c < 2 * C) ? f3 : f5;
  int cc = c % C;
  f[((size_t)b * HW + n) * 3 * C + c] = (_Float16)s[((size_t)b * C + cc) * HW + n];
}

__global__ void mulclip_kernel(const float* __restrict__ a, const float* __restrict__ b,
                               float* __restrict__ o, _Float16* __restrict__ oh, int n) {
  int i = blockIdx.x * blockDim.x + threadIdx.x;
  if (i < n) {
    float v = fminf(fmaxf(a[i] * b[i], -10.f), 10.f);
    o[i] = v;
    oh[i] = (_Float16)v;
  }
}

__global__ void gf_mean_kernel(const float* __restrict__ xm, float* __restrict__ gf,
                               int HW, int C) {
  int b = blockIdx.y, c = blockIdx.x;
  __shared__ float red[256];
  const float* p = xm + ((size_t)b * C + c) * HW;
  float s = 0.f;
  for (int i = threadIdx.x; i < HW; i += 256) s += p[i];
  red[threadIdx.x] = s;
  __syncthreads();
  for (int st = 128; st > 0; st >>= 1) {
    if (threadIdx.x < st) red[threadIdx.x] += red[threadIdx.x + st];
    __syncthreads();
  }
  if (threadIdx.x == 0) gf[b * C + c] = red[0] / (float)HW;
}

__global__ void fc_top2_kernel(const float* __restrict__ gf, const float* __restrict__ fcW,
                               const float* __restrict__ fcB, float* __restrict__ wts, int C) {
  int b = blockIdx.x;
  if (threadIdx.x != 0) return;
  float lg[4], mx = -3e38f;
  for (int e = 0; e < 4; ++e) {
    float s = fcB[e];
    for (int c = 0; c < C; ++c) s += fcW[e * C + c] * gf[b * C + c];
    lg[e] = s;
    mx = fmaxf(mx, s);
  }
  float z = 0.f;
  for (int e = 0; e < 4; ++e) { lg[e] = __expf(lg[e] - mx); z += lg[e]; }
  for (int e = 0; e < 4; ++e) lg[e] /= z;
  int i1 = 0;
  for (int e = 1; e < 4; ++e) if (lg[e] > lg[i1]) i1 = e;
  int i2 = (i1 == 0) ? 1 : 0;
  for (int e = 0; e < 4; ++e) if (e != i1 && lg[e] > lg[i2]) i2 = e;
  float s2 = lg[i1] + lg[i2] + 1e-8f;
  for (int e = 0; e < 4; ++e) wts[b * 4 + e] = 0.f;
  wts[b * 4 + i1] = lg[i1] / s2;
  wts[b * 4 + i2] = lg[i2] / s2;
}

__global__ void wcomb_kernel(const float* __restrict__ expW, const float* __restrict__ expB,
                             const float* __restrict__ wts, _Float16* __restrict__ wOut,
                             float* __restrict__ bOut, int C) {
  int b = blockIdx.y;
  int idx = blockIdx.x * blockDim.x + threadIdx.x;
  int sz = C * C * 9;
  if (idx < sz) {
    float s = 0.f;
    for (int e = 0; e < 4; ++e) s += wts[b * 4 + e] * expW[(size_t)e * sz + idx];
    wOut[(size_t)b * sz + idx] = (_Float16)s;
  }
  if (idx < C) {
    float s = 0.f;
    for (int e = 0; e < 4; ++e) s += wts[b * 4 + e] * expB[e * C + idx];
    bOut[b * C + idx] = s;
  }
}

// ---------------------------------------------------------------------------
// Host-side orchestration
// ---------------------------------------------------------------------------
static inline char* bump(char*& p, size_t bytes) {
  char* r = p;
  p += (bytes + 255) & ~(size_t)255;
  return r;
}

static void launch_gemm(hipStream_t st, const _Float16* A, int lda, int tA,
                        const _Float16* B, int ldb, int tB, float* oF, _Float16* oH,
                        int ldo, int tO, int M, int N, int K, const float* bias,
                        const float* g, const float* bb, int mode) {
  dim3 grid((M + 127) / 128, (N + 63) / 64);
  gemm_wmma_kernel<<<grid, 256, 0, st>>>(A, lda, tA, B, ldb, tB, oF, oH, ldo, tO,
                                         M, N, K, bias, g, bb, mode);
}

static void cast16(hipStream_t st, const float* s, _Float16* d, int n) {
  cast_f16_kernel<<<(n + 255) / 256, 256, 0, st>>>(s, d, n);
}

extern "C" void kernel_launch(void* const* d_in, const int* in_sizes, int n_in,
                              void* d_out, int out_size, void* d_ws, size_t ws_size,
                              hipStream_t stream) {
  (void)in_sizes; (void)n_in; (void)out_size; (void)ws_size;
  const int Bb = 2, C = 96, H = 64, W = 64, HW = 4096, K9 = C * 9;
  const int SC = 32, Nn = HW, TC = 3 * C, D64 = 64;

  const float* x     = (const float*)d_in[0];
  const float* cs_w  = (const float*)d_in[1];
  const float* cs_g  = (const float*)d_in[2];
  const float* cs_b  = (const float*)d_in[3];
  const float* dw_w  = (const float*)d_in[4];
  const float* dw_g  = (const float*)d_in[5];
  const float* dw_b  = (const float*)d_in[6];
  const float* proj_w= (const float*)d_in[7];
  const float* proj_b= (const float*)d_in[8];
  const float* br3_w = (const float*)d_in[9];
  const float* br3_b = (const float*)d_in[10];
  const float* br5_w = (const float*)d_in[11];
  const float* br5_b = (const float*)d_in[12];
  const float* br7_w = (const float*)d_in[13];
  const float* br7_b = (const float*)d_in[14];
  const float* bn_g  = (const float*)d_in[15];
  const float* bn_b  = (const float*)d_in[16];
  const float* exp_w = (const float*)d_in[17];
  const float* exp_b = (const float*)d_in[18];
  const float* fc_w  = (const float*)d_in[19];
  const float* fc_b  = (const float*)d_in[20];
  const float* m1_w  = (const float*)d_in[21];
  const float* m1_b  = (const float*)d_in[22];
  const float* m3_w  = (const float*)d_in[23];
  const float* m3_b  = (const float*)d_in[24];
  const float* m5_w  = (const float*)d_in[25];
  const float* m5_b  = (const float*)d_in[26];
  const float* q_w   = (const float*)d_in[27];
  const float* q_b   = (const float*)d_in[28];
  const float* k_w   = (const float*)d_in[29];
  const float* k_b   = (const float*)d_in[30];
  const float* v_w   = (const float*)d_in[31];
  const float* v_b   = (const float*)d_in[32];
  const float* o_w   = (const float*)d_in[33];
  const float* o_b   = (const float*)d_in[34];
  const float* ot1_w = (const float*)d_in[35];
  const float* ot1_g = (const float*)d_in[36];
  const float* ot1_b = (const float*)d_in[37];
  const float* ot2_w = (const float*)d_in[38];
  const float* ot2_g = (const float*)d_in[39];
  const float* ot2_b = (const float*)d_in[40];
  float* out = (float*)d_out;

  // ----- workspace layout -----
  char* p = (char*)d_ws;
  const size_t BCHW = (size_t)Bb * C * HW;
  float* x1      = (float*)bump(p, BCHW * 4);
  float* xs      = (float*)bump(p, BCHW * 4);
  float* sp      = (float*)bump(p, BCHW * 4);
  float* xmulti  = (float*)bump(p, BCHW * 4);
  float* outdt   = (float*)bump(p, BCHW * 4);
  float* f1      = (float*)bump(p, BCHW * 4);
  float* f3      = (float*)bump(p, BCHW * 4);
  float* f5      = (float*)bump(p, BCHW * 4);
  float* outmdam = (float*)bump(p, BCHW * 4);
  float* x2      = (float*)bump(p, BCHW * 4);
  float* x3      = (float*)bump(p, BCHW * 4);
  float* gf      = (float*)bump(p, (size_t)Bb * C * 4);
  float* wts     = (float*)bump(p, (size_t)Bb * 4 * 4);
  float* rowmax  = (float*)bump(p, (size_t)Bb * Nn * 4);
  float* rowsum  = (float*)bump(p, (size_t)Bb * Nn * 4);
  float* keyimp  = (float*)bump(p, (size_t)Bb * Nn * 4);
  float* kmA     = (float*)bump(p, (size_t)Bb * Nn * 4);
  float* bcomb   = (float*)bump(p, (size_t)Bb * C * 4);
  _Float16* cs_w16   = (_Float16*)bump(p, (size_t)C * K9 * 2);
  _Float16* wcomb16  = (_Float16*)bump(p, (size_t)Bb * C * K9 * 2);
  _Float16* proj_w16 = (_Float16*)bump(p, (size_t)3 * SC * SC * 2);
  _Float16* m1_w16   = (_Float16*)bump(p, (size_t)C * C * 2);
  _Float16* q_w16    = (_Float16*)bump(p, (size_t)D64 * TC * 2);
  _Float16* k_w16    = (_Float16*)bump(p, (size_t)D64 * TC * 2);
  _Float16* v_w16    = (_Float16*)bump(p, (size_t)C * TC * 2);
  _Float16* o_w16    = (_Float16*)bump(p, (size_t)C * C * 2);
  _Float16* ot1_w16  = (_Float16*)bump(p, (size_t)C * C * 2);
  _Float16* xs16     = (_Float16*)bump(p, BCHW * 2);
  _Float16* col16    = (_Float16*)bump(p, (size_t)HW * K9 * 2);
  _Float16* fbuf     = (_Float16*)bump(p, (size_t)Bb * HW * TC * 2);
  _Float16* Q16      = (_Float16*)bump(p, (size_t)Bb * Nn * D64 * 2);
  _Float16* K16      = (_Float16*)bump(p, (size_t)Bb * Nn * D64 * 2);
  _Float16* V16      = (_Float16*)bump(p, (size_t)Bb * Nn * C * 2);
  _Float16* ao16     = (_Float16*)bump(p, (size_t)Bb * Nn * C * 2);
  _Float16* x2_16    = (_Float16*)bump(p, BCHW * 2);

  // ----- weight fp16 casts -----
  cast16(stream, cs_w,  cs_w16,  C * K9);
  cast16(stream, proj_w, proj_w16, 3 * SC * SC);
  cast16(stream, m1_w,  m1_w16,  C * C);
  cast16(stream, q_w,   q_w16,   D64 * TC);
  cast16(stream, k_w,   k_w16,   D64 * TC);
  cast16(stream, v_w,   v_w16,   C * TC);
  cast16(stream, o_w,   o_w16,   C * C);
  cast16(stream, ot1_w, ot1_w16, C * C);

  const int im2colBlocks = (HW * K9 + 255) / 256;

  // ----- conv_start (im2col + WMMA GEMM, bn+relu epilogue) -----
  for (int b = 0; b < Bb; ++b) {
    im2col3_kernel<<<im2colBlocks, 256, 0, stream>>>(x + (size_t)b * C * HW, col16, C, H, W);
    launch_gemm(stream, col16, K9, 0, cs_w16, K9, 0, x1 + (size_t)b * C * HW, nullptr,
                HW, 1, HW, C, K9, nullptr, cs_g, cs_b, 2);
  }
  // ----- dw_shared: relu(bn(dwconv)) -----
  for (int b = 0; b < Bb; ++b)
    dwconv_kernel<<<dim3(16, C), 256, 0, stream>>>(x1 + (size_t)b * C * HW,
        xs + (size_t)b * C * HW, dw_w, nullptr, dw_g, dw_b, nullptr, H, W, 3, 1);
  cast16(stream, xs, xs16, (int)BCHW);

  // ----- DTConv: proj einsum (6 small WMMA GEMMs) -----
  for (int b = 0; b < Bb; ++b)
    for (int s = 0; s < 3; ++s)
      launch_gemm(stream, xs16 + ((size_t)b * C + s * SC) * HW, HW, 1,
                  proj_w16 + (size_t)s * SC * SC, SC, 0,
                  sp + ((size_t)b * C + s * SC) * HW, nullptr, HW, 1,
                  HW, SC, SC, proj_b + s * SC, nullptr, nullptr, 1);
  // branches: bn(relu(dwconv+bias))
  const int kss[3] = {3, 5, 7};
  const float* brw[3] = {br3_w, br5_w, br7_w};
  const float* brb[3] = {br3_b, br5_b, br7_b};
  for (int b = 0; b < Bb; ++b)
    for (int s = 0; s < 3; ++s)
      dwconv_kernel<<<dim3(16, SC), 256, 0, stream>>>(
          sp + ((size_t)b * C + s * SC) * HW, xmulti + ((size_t)b * C + s * SC) * HW,
          brw[s], brb[s], bn_g + s * SC, bn_b + s * SC, nullptr, H, W, kss[s], 2);
  // gating
  gf_mean_kernel<<<dim3(C, Bb), 256, 0, stream>>>(xmulti, gf, HW, C);
  fc_top2_kernel<<<Bb, 32, 0, stream>>>(gf, fc_w, fc_b, wts, C);
  wcomb_kernel<<<dim3((C * C * 9 + 255) / 256, Bb), 256, 0, stream>>>(
      exp_w, exp_b, wts, wcomb16, bcomb, C);
  // expert conv folded into one GEMM per batch
  for (int b = 0; b < Bb; ++b) {
    im2col3_kernel<<<im2colBlocks, 256, 0, stream>>>(xs + (size_t)b * C * HW, col16, C, H, W);
    launch_gemm(stream, col16, K9, 0, wcomb16 + (size_t)b * C * K9, K9, 0,
                outdt + (size_t)b * C * HW, nullptr, HW, 1, HW, C, K9,
                bcomb + b * C, nullptr, nullptr, 1);
  }

  // ----- MDAM: f1 (1x1 WMMA), f3/f5 (dw), concat to fp16 -----
  for (int b = 0; b < Bb; ++b) {
    launch_gemm(stream, xs16 + (size_t)b * C * HW, HW, 1, m1_w16, C, 0,
                f1 + (size_t)b * C * HW, nullptr, HW, 1, HW, C, C, m1_b, nullptr, nullptr, 1);
    dwconv_kernel<<<dim3(16, C), 256, 0, stream>>>(xs + (size_t)b * C * HW,
        f3 + (size_t)b * C * HW, m3_w, m3_b, nullptr, nullptr, nullptr, H, W, 3, 0);
    dwconv_kernel<<<dim3(16, C), 256, 0, stream>>>(xs + (size_t)b * C * HW,
        f5 + (size_t)b * C * HW, m5_w, m5_b, nullptr, nullptr, nullptr, H, W, 5, 0);
  }
  gather_f_kernel<<<dim3((HW * TC + 255) / 256, Bb), 256, 0, stream>>>(f1, f3, f5, fbuf, C, HW);

  // Q/K/V projections (WMMA, fp16 outputs)
  launch_gemm(stream, fbuf, TC, 0, q_w16, TC, 0, nullptr, Q16, D64, 0,
              Bb * HW, D64, TC, q_b, nullptr, nullptr, 1);
  launch_gemm(stream, fbuf, TC, 0, k_w16, TC, 0, nullptr, K16, D64, 0,
              Bb * HW, D64, TC, k_b, nullptr, nullptr, 1);
  launch_gemm(stream, fbuf, TC, 0, v_w16, TC, 0, nullptr, V16, C, 0,
              Bb * HW, C, TC, v_b, nullptr, nullptr, 1);

  // streaming attention
  zero_kernel<<<(Bb * Nn + 255) / 256, 256, 0, stream>>>(keyimp, Bb * Nn);
  attn_pass1_kernel<<<dim3(Nn / 64, Bb), 256, 0, stream>>>(Q16, K16, rowmax, rowsum, Nn);
  attn_colsum_kernel<<<dim3(Nn / 64, Bb), 256, 0, stream>>>(Q16, K16, rowmax, rowsum, keyimp, Nn);
  topk_mask_kernel<<<Bb, 1024, 0, stream>>>(keyimp, kmA, Nn, Nn / 2);
  attn_pass2_kernel<<<dim3(Nn / 64, Bb), 256, 0, stream>>>(Q16, K16, V16, rowmax, rowsum,
                                                           kmA, ao16, Nn);
  // output projection -> out_mdam [C, HW]
  for (int b = 0; b < Bb; ++b)
    launch_gemm(stream, ao16 + (size_t)b * Nn * C, C, 0, o_w16, C, 0,
                outmdam + (size_t)b * C * HW, nullptr, HW, 1, HW, C, C,
                o_b, nullptr, nullptr, 1);

  // ----- combine + output_trans + residual -----
  mulclip_kernel<<<(int)((BCHW + 255) / 256), 256, 0, stream>>>(outdt, outmdam, x2, x2_16, (int)BCHW);
  for (int b = 0; b < Bb; ++b)
    launch_gemm(stream, x2_16 + (size_t)b * C * HW, HW, 1, ot1_w16, C, 0,
                x3 + (size_t)b * C * HW, nullptr, HW, 1, HW, C, C,
                nullptr, ot1_g, ot1_b, 2);
  for (int b = 0; b < Bb; ++b)
    dwconv_kernel<<<dim3(16, C), 256, 0, stream>>>(x3 + (size_t)b * C * HW,
        out + (size_t)b * C * HW, ot2_w, nullptr, ot2_g, ot2_b,
        x + (size_t)b * C * HW, H, W, 3, 3);
}
